// DiscreteGaugeConnection_81252191306167
// MI455X (gfx1250) — compile-verified, matrix operand order checked by reading, not run on hardware
//
// MI455X / gfx1250: fused MLP (WMMA f16 -> f32) + per-lane 8x8 skew expm.
//
// Roofline: 144 MB traffic (~6us @ 23.3 TB/s) vs ~5 GFLOP fp32 expm -> compute
// bound in expm; MLP GEMMs (M=512K, K<=32, N<=32) mapped onto
// v_wmma_f32_16x16x32_f16 (16-row tiles per wave, K zero-padded for layer 1).
// Layer1->Layer2 layout transpose staged through per-wave LDS (stride-40
// halves => 16B-aligned ds_load_b128; same-wave DS ops are in-order, no
// barrier needed). Omega params staged with stride-33 floats (odd stride =>
// conflict-free per-lane matrix reads). expm: scaling & squaring, Taylor-6
// Horner, all 8x8 state in VGPRs (constant-index arrays), rolled loops to
// bound I-cache footprint. tanh is branch-free (v_exp_f32 + v_rcp_f32) to
// avoid the OCML exec-mask-diverged path seen in round 1.

#include <hip/hip_runtime.h>
#include <hip/hip_bf16.h>

typedef __attribute__((ext_vector_type(16))) _Float16 v16h;
typedef __attribute__((ext_vector_type(8)))  _Float16 v8h;
typedef __attribute__((ext_vector_type(8)))  float    v8f;

#define DIM 8
#define HID 32
#define NT  28   // 8*7/2 upper-triangular params

__device__ __forceinline__ v8f wmma_f16_f32(v16h a, v16h b) {
  v8f c = {0.f, 0.f, 0.f, 0.f, 0.f, 0.f, 0.f, 0.f};
  // 8 args: (neg_a, A, neg_b, B, c_mod, C, reuse_a, reuse_b)
  return __builtin_amdgcn_wmma_f32_16x16x32_f16(false, a, false, b,
                                                (short)0, c, false, false);
}

// Branch-free tanh: tanh(x) = 1 - 2/(exp2(2*log2e*x)+1).
// t->inf => rcp->0 => +1 ; t->0 => -1. ~1 ulp of HW exp, far below the f16
// quantization applied to h afterwards. 4 VALU ops, no EXEC divergence.
__device__ __forceinline__ float fast_tanh(float x) {
  float t = __builtin_amdgcn_exp2f(x * 2.8853900817779268f);
  return 1.f - 2.f * __builtin_amdgcn_rcpf(t + 1.f);
}

// Fully register-resident 8x8 matmul (constant indices only).
__device__ __forceinline__ void mm8(const float (&A)[8][8],
                                    const float (&B)[8][8],
                                    float (&C)[8][8]) {
#pragma unroll
  for (int i = 0; i < 8; ++i) {
#pragma unroll
    for (int j = 0; j < 8; ++j) {
      float acc = 0.f;
#pragma unroll
      for (int k = 0; k < 8; ++k)
        acc = __builtin_fmaf(A[i][k], B[k][j], acc);
      C[i][j] = acc;
    }
  }
}

__global__ __launch_bounds__(256, 1)
void gauge_expm_kernel(const float* __restrict__ x,
                       const float* __restrict__ W1,
                       const float* __restrict__ b1,
                       const float* __restrict__ W2,
                       const float* __restrict__ b2,
                       float* __restrict__ out,
                       int Mtot) {
  // Per-wave LDS regions (8 waves per 256-thread block):
  // h stage: 16 rows x 32 hidden (f16), row stride 40 halves (80B, 16B mult)
  // omega stage: 32 rows x 28 params (f32), row stride 33 (odd -> no conflicts)
  __shared__ __align__(16) _Float16 hstage[8][16 * 40];
  __shared__ float wst[8][32 * 33];

  const int tid  = threadIdx.x;
  const int wid  = tid >> 5;
  const int lane = tid & 31;
  const int n0   = lane & 15;          // column index within 16-wide fragment
  const int kb   = (lane >> 4) << 3;   // 0 or 8: K-half selector (wave32 layout)

  // ---- Build B fragments for W1 (8x32, K-padded to 32) and W2 (32x28->32) ---
  // 16-bit B layout (32x16 K x N): lane = N (mod 16); halves 0..7 -> K=kb..kb+7,
  // halves 8..15 -> K=16+kb..23+kb.
  v16h bW1[2], bW2[2];
  float bias1[2], bias2[2];
#pragma unroll
  for (int j = 0; j < 2; ++j) {
    const int N = n0 + 16 * j;
#pragma unroll
    for (int h = 0; h < 16; ++h) {
      const int K = (h < 8 ? h : h + 8) + kb;
      bW1[j][h] = (K < DIM) ? (_Float16)W1[K * HID + N] : (_Float16)0.f;
      bW2[j][h] = (N < NT)  ? (_Float16)W2[K * NT + N]  : (_Float16)0.f;
    }
    bias1[j] = b1[N];
    bias2[j] = (N < NT) ? b2[N] : 0.f;
  }

  const int base32 = blockIdx.x * 256 + wid * 32;  // 32 rows per wave

  // ---- Two 16-row WMMA tiles per wave --------------------------------------
#pragma unroll
  for (int t = 0; t < 2; ++t) {
    const int rowbase = base32 + t * 16;

    // A fragment for layer 1: lanes 0..15 carry K=0..7 (halves 0..7); all
    // other halves (K=8..31) are the zero padding.
    v16h a1;
#pragma unroll
    for (int h = 0; h < 16; ++h) a1[h] = (_Float16)0.f;
    if (lane < 16) {
      long gm = (long)rowbase + n0;
      if (gm > (long)Mtot - 1) gm = (long)Mtot - 1;
      const float* xr = x + gm * (long)DIM;
#pragma unroll
      for (int k = 0; k < 8; ++k) a1[k] = (_Float16)xr[k];
    }

    v8f h0 = wmma_f16_f32(a1, bW1[0]);
    v8f h1 = wmma_f16_f32(a1, bW1[1]);

    // C layout: lane n0 = column, reg r -> row r + 8*(lane>>4).
    const int mloc = (lane >> 4) * 8;
#pragma unroll
    for (int r = 0; r < 8; ++r) {
      hstage[wid][(mloc + r) * 40 + n0]      = (_Float16)fast_tanh(h0[r] + bias1[0]);
      hstage[wid][(mloc + r) * 40 + n0 + 16] = (_Float16)fast_tanh(h1[r] + bias1[1]);
    }

    // Re-read in 16-bit A layout: lane -> row M=n0; two aligned b128 reads.
    const _Float16* hs = &hstage[wid][0];
    v8h lo = *(const v8h*)(hs + n0 * 40 + kb);        // K = kb..kb+7
    v8h hi = *(const v8h*)(hs + n0 * 40 + 16 + kb);   // K = 16+kb..23+kb
    v16h a2;
#pragma unroll
    for (int i = 0; i < 8; ++i) { a2[i] = lo[i]; a2[8 + i] = hi[i]; }

    v8f o0 = wmma_f16_f32(a2, bW2[0]);
    v8f o1 = wmma_f16_f32(a2, bW2[1]);

    const int mo = mloc + 16 * t;
#pragma unroll
    for (int r = 0; r < 8; ++r) {
      wst[wid][(mo + r) * 33 + n0] = o0[r] + bias2[0];
      if (n0 < NT - 16)
        wst[wid][(mo + r) * 33 + n0 + 16] = o1[r] + bias2[1];
    }
  }

  // ---- Per-lane expm of its own 8x8 skew-symmetric matrix ------------------
  float p[NT];
#pragma unroll
  for (int n = 0; n < NT; ++n) p[n] = wst[wid][lane * 33 + n];

  float X[8][8], T[8][8], U[8][8];
  {
    int idx = 0;
#pragma unroll
    for (int i = 0; i < 8; ++i) {
      X[i][i] = 0.f;
#pragma unroll
      for (int j = i + 1; j < 8; ++j) {
        X[i][j] = p[idx];
        X[j][i] = -p[idx];
        ++idx;
      }
    }
  }

  // Scaling: s such that ||X||_F / 2^s <= 0.25 (bit tricks, no libm).
  float fro2 = 0.f;
#pragma unroll
  for (int n = 0; n < NT; ++n) fro2 = __builtin_fmaf(p[n], p[n], fro2);
  const float nrm = __builtin_sqrtf(2.f * fro2);
  int s = 0;
  if (nrm > 0.25f) {
    s = (int)((__float_as_uint(nrm) >> 23) & 0xff) - 124;  // E+3
    if (s < 0) s = 0;
    if (s > 30) s = 30;
  }
  const float sc = __int_as_float((127 - s) << 23);  // 2^-s
#pragma unroll
  for (int i = 0; i < 8; ++i)
#pragma unroll
    for (int j = 0; j < 8; ++j) X[i][j] *= sc;

  // Taylor-6 via Horner: T = I; T = I + (X@T)/k for k = 6..1.
#pragma unroll
  for (int i = 0; i < 8; ++i)
#pragma unroll
    for (int j = 0; j < 8; ++j) T[i][j] = (i == j) ? 1.f : 0.f;
#pragma unroll 1
  for (int k = 6; k >= 1; --k) {
    mm8(X, T, U);
    const float invk = 1.f / (float)k;
#pragma unroll
    for (int i = 0; i < 8; ++i)
#pragma unroll
      for (int j = 0; j < 8; ++j)
        T[i][j] = ((i == j) ? 1.f : 0.f) + U[i][j] * invk;
  }

  // s squarings (per-lane trip count; EXEC-masked divergence is fine).
#pragma unroll 1
  for (int q = 0; q < s; ++q) {
    mm8(T, T, U);
#pragma unroll
    for (int i = 0; i < 8; ++i)
#pragma unroll
      for (int j = 0; j < 8; ++j) T[i][j] = U[i][j];
  }

  // ---- Store: lane owns 64 contiguous floats -> 16x global_store_b128 ------
  const long gm = (long)base32 + lane;
  if (gm < (long)Mtot) {
    float4* o = (float4*)(out + gm * 64);
#pragma unroll
    for (int i = 0; i < 8; ++i) {
      o[2 * i]     = make_float4(T[i][0], T[i][1], T[i][2], T[i][3]);
      o[2 * i + 1] = make_float4(T[i][4], T[i][5], T[i][6], T[i][7]);
    }
  }
}

extern "C" void kernel_launch(void* const* d_in, const int* in_sizes, int n_in,
                              void* d_out, int out_size, void* d_ws, size_t ws_size,
                              hipStream_t stream) {
  (void)n_in; (void)out_size; (void)d_ws; (void)ws_size;
  const float* x  = (const float*)d_in[0];   // diff_vec (B,N,N,8) f32
  const float* W1 = (const float*)d_in[1];   // (8,32)
  const float* b1 = (const float*)d_in[2];   // (32,)
  const float* W2 = (const float*)d_in[3];   // (32,28)
  const float* b2 = (const float*)d_in[4];   // (28,)
  float* out = (float*)d_out;                // (...,8,8) f32

  const int Mtot = in_sizes[0] / DIM;        // 524288 for the given shapes
  const int blocks = (Mtot + 255) / 256;     // 256 rows per block (8 waves x 32)
  gauge_expm_kernel<<<blocks, 256, 0, stream>>>(x, W1, b1, W2, b2, out, Mtot);
}
